// NXTentLoss_72619307041593
// MI455X (gfx1250) — matrix operand order repaired and checked by reading, not compiled, not run on hardware
//
#include <hip/hip_runtime.h>
#include <hip/hip_bf16.h>
#include <math.h>

typedef _Float16 half_t;
typedef __attribute__((ext_vector_type(16))) _Float16 v16h;
typedef __attribute__((ext_vector_type(8)))  _Float16 v8h;
typedef __attribute__((ext_vector_type(4)))  _Float16 v4h;
typedef __attribute__((ext_vector_type(8)))  float    v8f;

#define DIM 128          // feature dimension (row stride)
#define MT  4            // 16-row tiles per block (64 rows/block)
#define TWO_OVER_LN2 2.8853900817779268f  // exp(2x) == exp2(x * 2/ln2)

__device__ __forceinline__ v16h cat8(v8h a, v8h b) {
  return __builtin_shufflevector(a, b, 0, 1, 2, 3, 4, 5, 6, 7, 8, 9, 10, 11, 12, 13, 14, 15);
}

// ---------------------------------------------------------------------------
// Kernel 1: row-normalize fp32 -> f16. One wave (32 lanes) per row of 128.
// ---------------------------------------------------------------------------
__global__ __launch_bounds__(256) void nxt_normalize(const float* __restrict__ x,
                                                     half_t* __restrict__ z, int rows) {
  const int row  = (int)((blockIdx.x * blockDim.x + threadIdx.x) >> 5);
  const int lane = threadIdx.x & 31;
  if (row >= rows) return;

  const float4 v = *(const float4*)(x + (size_t)row * DIM + lane * 4);
  float ss = v.x * v.x + v.y * v.y + v.z * v.z + v.w * v.w;
#pragma unroll
  for (int off = 16; off >= 1; off >>= 1) ss += __shfl_xor(ss, off, 32);

  const float inv = 1.0f / fmaxf(sqrtf(ss), 1e-6f);  // torch-style eps clamp
  v4h o;
  o[0] = (half_t)(v.x * inv);
  o[1] = (half_t)(v.y * inv);
  o[2] = (half_t)(v.z * inv);
  o[3] = (half_t)(v.w * inv);
  *(v4h*)(z + (size_t)row * DIM + lane * 4) = o;
}

// ---------------------------------------------------------------------------
// Kernel 2: fused sim-GEMM + exp + row reduction.
// Block owns 4 consecutive 16-row tiles (A resident, 128 VGPRs); 8 waves each
// sweep every 8th 16-col tile. Per B fragment: 16 chained v_wmma_f32_16x16x32_f16.
// Steady-state per element: v_pk_mul + v_exp_f32 + v_pk_add. Diagonal exclusion
// and positive-pair capture sit behind wave-uniform branches (<=1/64 iterations).
// ---------------------------------------------------------------------------
__global__ __launch_bounds__(256) void nxt_sim_loss(const half_t* __restrict__ z,
                                                    float* __restrict__ blockloss,
                                                    int ntiles) {
  const int it0    = blockIdx.x * MT;
  const int wave   = threadIdx.x >> 5;
  const int lane   = threadIdx.x & 31;
  const int lane16 = lane & 15;
  const int hi     = lane >> 4;  // 0: lanes 0-15, 1: lanes 16-31

  __shared__ float denom_s[8][MT][16];
  __shared__ float pos_s[MT][16];

  // --- A fragments: rows it0*16 .. it0*16+63, 4 K-blocks each, kept resident.
  // A layout: lane holds row lane16; hi=0 -> K {k0..k0+7, k0+16..k0+23}, hi=1 -> +8.
  v16h a[MT][4];
#pragma unroll
  for (int mt = 0; mt < MT; ++mt) {
    const half_t* ap = z + (size_t)((it0 + mt) * 16 + lane16) * DIM + hi * 8;
#pragma unroll
    for (int kb = 0; kb < 4; ++kb) {
      v8h lo = *(const v8h*)(ap + kb * 32);
      v8h hh = *(const v8h*)(ap + kb * 32 + 16);
      a[mt][kb] = cat8(lo, hh);
    }
  }

  float acc[MT][8];
  float posv[MT];
  int   jtp[MT];
#pragma unroll
  for (int mt = 0; mt < MT; ++mt) {
    posv[mt] = 0.0f;
    jtp[mt]  = (it0 + mt + (ntiles >> 1)) % ntiles;  // tile holding positives on its diag
#pragma unroll
    for (int r = 0; r < 8; ++r) acc[mt][r] = 0.0f;
  }

  for (int jt = wave; jt < ntiles; jt += 8) {
    // --- B fragment: lane holds column lane16 (== Z row jt*16+lane16);
    //     hi=0 -> K = kb*32 + {0..7, 16..23}, hi=1 -> K = kb*32 + {8..15, 24..31}.
    const half_t* bp = z + (size_t)(jt * 16 + lane16) * DIM + hi * 8;
    // Warm the NEAR cache for the next iteration's tile: locality 3 -> WGP scope
    // (locality 0 lowered to SCOPE_SYS, which skips the WGP prefetch entirely).
    __builtin_prefetch((const void*)(bp + 8 * 16 * DIM), 0, 3);

    v16h b[4];
#pragma unroll
    for (int kb = 0; kb < 4; ++kb) {
      v8h lo = *(const v8h*)(bp + kb * 32);
      v8h hh = *(const v8h*)(bp + kb * 32 + 16);
      b[kb] = cat8(lo, hh);
    }

#pragma unroll
    for (int mt = 0; mt < MT; ++mt) {
      v8f c = {};
#pragma unroll
      for (int kb = 0; kb < 4; ++kb)
        c = __builtin_amdgcn_wmma_f32_16x16x32_f16(false, a[mt][kb], false, b[kb],
                                                   (short)0, c, false, false);
      // Steady state: unconditional exp + accumulate (no per-element selects).
      float e[8];
#pragma unroll
      for (int r = 0; r < 8; ++r) {
        e[r] = __builtin_amdgcn_exp2f(c[r] * TWO_OVER_LN2);  // exp(sim / 0.5)
        acc[mt][r] += e[r];
      }
      // Cold paths: wave-uniform branches, each taken at most once per wave.
      if (jt == it0 + mt) {               // tile containing the self-similarity diag
#pragma unroll
        for (int r = 0; r < 8; ++r) {
          const bool idiag = ((r + hi * 8) == lane16);
          acc[mt][r] -= idiag ? e[r] : 0.0f;  // denom excludes self-sim
        }
      }
      if (jt == jtp[mt]) {                // tile whose diagonal holds positives
#pragma unroll
        for (int r = 0; r < 8; ++r) {
          const bool idiag = ((r + hi * 8) == lane16);
          posv[mt] = idiag ? e[r] : posv[mt];
        }
      }
    }
  }

  // One predicated LDS store per row-tile for the positive values.
  const bool posLane = ((lane16 >> 3) == hi);  // lane that held the in-tile diagonal
#pragma unroll
  for (int mt = 0; mt < MT; ++mt) {
    if (((jtp[mt] & 7) == wave) && posLane) pos_s[mt][lane16] = posv[mt];
  }

  // Reduce row partials across the 16 lanes of each half-wave.
#pragma unroll
  for (int off = 1; off <= 8; off <<= 1) {
#pragma unroll
    for (int mt = 0; mt < MT; ++mt)
#pragma unroll
      for (int r = 0; r < 8; ++r) acc[mt][r] += __shfl_xor(acc[mt][r], off, 32);
  }
  if (lane16 == 0) {
#pragma unroll
    for (int mt = 0; mt < MT; ++mt)
#pragma unroll
      for (int r = 0; r < 8; ++r) denom_s[wave][mt][r + hi * 8] = acc[mt][r];
  }
  __syncthreads();

  // Wave 0: combine the 8 wave-partials per row; 64 rows -> 2 per lane.
  if (wave == 0) {
    float v = 0.0f;
#pragma unroll
    for (int halfidx = 0; halfidx < 2; ++halfidx) {
      const int R   = lane + halfidx * 32;  // 0..63
      const int mt  = R >> 4;
      const int row = R & 15;
      float denom = 0.0f;
#pragma unroll
      for (int w = 0; w < 8; ++w) denom += denom_s[w][mt][row];
      // -log(pos / sqrt(denom)) = -log(pos) + 0.5*log(denom)
      v += -__logf(pos_s[mt][row]) + 0.5f * __logf(denom);
    }
#pragma unroll
    for (int off = 16; off >= 1; off >>= 1) v += __shfl_xor(v, off, 32);
    if (lane == 0) blockloss[blockIdx.x] = v;
  }
}

// ---------------------------------------------------------------------------
// Kernel 3: reduce per-block partials to the scalar loss.
// ---------------------------------------------------------------------------
__global__ __launch_bounds__(32) void nxt_reduce(const float* __restrict__ bl,
                                                 float* __restrict__ out, int n) {
  const int lane = threadIdx.x;
  float s = 0.0f;
  for (int i = lane; i < n; i += 32) s += bl[i];
#pragma unroll
  for (int off = 16; off >= 1; off >>= 1) s += __shfl_xor(s, off, 32);
  if (lane == 0) out[0] = s;
}

extern "C" void kernel_launch(void* const* d_in, const int* in_sizes, int n_in,
                              void* d_out, int out_size, void* d_ws, size_t ws_size,
                              hipStream_t stream) {
  const float* x   = (const float*)d_in[0];
  const int rows   = in_sizes[0] / DIM;  // 8192
  const int ntiles = rows / 16;          // 512
  const int nblk   = ntiles / MT;        // 128

  half_t* z  = (half_t*)d_ws;                                          // 2 MB f16 normalized
  float*  bl = (float*)((char*)d_ws + (size_t)rows * DIM * sizeof(half_t));

  nxt_normalize<<<(rows + 7) / 8, 256, 0, stream>>>(x, z, rows);
  nxt_sim_loss<<<nblk, 256, 0, stream>>>(z, bl, ntiles);
  nxt_reduce<<<1, 32, 0, stream>>>(bl, (float*)d_out, nblk);
}
